// BiLSTM_Parser_simple_12249246728935
// MI455X (gfx1250) — matrix operand order repaired and checked by reading, not compiled, not run on hardware
//
#include <hip/hip_runtime.h>
#include <hip/hip_bf16.h>

// ---------------------------------------------------------------------------
// BiLSTM parser for MI455X (gfx1250, wave32, WMMA).
// All GEMMs (input projections, head/modif projections) and the recurrent
// h@Whh.T use v_wmma_f32_16x16x32_bf16. Recurrence: persistent kernel, one
// workgroup per direction, ping-pong LDS h-state, cell state c in registers.
// Batched GEMM is register-blocked 1 M-tile x 4 N-tiles per wave so each
// A-fragment load feeds 4 WMMAs.
// ---------------------------------------------------------------------------

typedef __attribute__((ext_vector_type(16))) __bf16 v16bf;
typedef __attribute__((ext_vector_type(8)))  float  v8f;

#define TT 128      // sequence length
#define BB 32       // batch
#define HH 400      // hidden per direction
#define G4 1600     // 4*H
#define KP0 416     // 400 padded to multiple of 32
#define LH 800      // 2*H
#define MHH 512     // MH/2
#define ROWS 4096   // B*T

__device__ __forceinline__ float sigf(float x) { return 1.0f / (1.0f + __expf(-x)); }

// A-fragment: 16x32 bf16 tile (M x K), row-major source, per ISA lane layout:
// lane L holds row m0+(L&15); lanes 0-15 K = k0+{0..7,16..23}, lanes 16-31 +8.
__device__ __forceinline__ v16bf frag_A(const __bf16* A, int lda, int m0, int k0) {
  const int lane = threadIdx.x & 31;
  const __bf16* p = A + (size_t)(m0 + (lane & 15)) * lda + k0 + ((lane >> 4) << 3);
  v16bf a;
#pragma unroll
  for (int j = 0; j < 8; ++j) a[j] = p[j];
#pragma unroll
  for (int j = 0; j < 8; ++j) a[8 + j] = p[16 + j];
  return a;
}

// B-fragment: 32x16 bf16 tile (K x N) taken from weight matrix W stored
// [N][ldb] row-major (i.e. B = W^T). lane L holds col n0+(L&15);
// lanes 0-15 K = k0+{0..7,16..23}, lanes 16-31 +8.
__device__ __forceinline__ v16bf frag_B(const __bf16* W, int ldb, int n0, int k0) {
  const int lane = threadIdx.x & 31;
  const __bf16* p = W + (size_t)(n0 + (lane & 15)) * ldb + k0 + ((lane >> 4) << 3);
  v16bf b;
#pragma unroll
  for (int j = 0; j < 8; ++j) b[j] = p[j];
#pragma unroll
  for (int j = 0; j < 8; ++j) b[8 + j] = p[16 + j];
  return b;
}

// ---------------- weight fp32 -> bf16 with K padding --------------------
__global__ void cvt_pad(const float* __restrict__ src, __bf16* __restrict__ dst,
                        int K, int Kpad) {
  const int row = blockIdx.x;
  for (int j = threadIdx.x; j < Kpad; j += blockDim.x)
    dst[(size_t)row * Kpad + j] = (j < K) ? (__bf16)src[(size_t)row * K + j] : (__bf16)0.0f;
}

// ---------------- embedding gather + concat -> bf16 ---------------------
__global__ void embed_concat(const int* __restrict__ wt, const int* __restrict__ pt,
                             const float* __restrict__ we, const float* __restrict__ pe,
                             __bf16* __restrict__ x0) {
  const int row = blockIdx.x;            // b*T + t
  const int widx = wt[row];
  const int pidx = pt[row];
  __bf16* dst = x0 + (size_t)row * KP0;
  for (int j = threadIdx.x; j < KP0; j += blockDim.x) {
    float v = 0.0f;
    if (j < 300)      v = we[(size_t)widx * 300 + j];
    else if (j < 400) v = pe[(size_t)pidx * 100 + (j - 300)];
    dst[j] = (__bf16)v;
  }
}

// ---------------- bf16 WMMA GEMM: C = A * W^T + bias --------------------
// A: [M][lda] bf16, W: [N][ldb] bf16, C: [M][ldc] f32.
// One wave per 16x64 output strip (1 M-tile x 4 N-tiles); each A fragment
// feeds 4 WMMAs. Requires N % 64 == 0, K % 32 == 0, M % 16 == 0.
__global__ void __launch_bounds__(256)
gemm_bf16_wmma(const __bf16* __restrict__ A, int lda,
               const __bf16* __restrict__ W, int ldb,
               const float* __restrict__ bias,
               float* __restrict__ C, int ldc,
               int M, int N, int K) {
  const int lane = threadIdx.x & 31;
  const int strip = blockIdx.x * (blockDim.x >> 5) + (threadIdx.x >> 5);
  const int nstrips = N >> 6;           // strips of 4 n-tiles
  const int mt = strip / nstrips;
  const int ns = strip % nstrips;
  const int m0 = mt << 4, n0 = ns << 6;
  if (m0 >= M) return;

  v8f acc[4];
#pragma unroll
  for (int q = 0; q < 4; ++q) {
    const float bv = bias[n0 + q * 16 + (lane & 15)];
#pragma unroll
    for (int r = 0; r < 8; ++r) acc[q][r] = bv;
  }

  for (int k0 = 0; k0 < K; k0 += 32) {
    v16bf a = frag_A(A, lda, m0, k0);
#pragma unroll
    for (int q = 0; q < 4; ++q) {
      v16bf b = frag_B(W, ldb, n0 + q * 16, k0);
      acc[q] = __builtin_amdgcn_wmma_f32_16x16x32_bf16(false, a, false, b,
                                                       (short)0, acc[q], false, false);
    }
  }
#pragma unroll
  for (int q = 0; q < 4; ++q)
#pragma unroll
    for (int r = 0; r < 8; ++r) {
      const int m = m0 + r + ((lane >> 4) << 3);
      C[(size_t)m * ldc + n0 + q * 16 + (lane & 15)] = acc[q][r];
    }
}

// ---------------- persistent bidirectional LSTM layer -------------------
// grid = 2 (direction), block = 512 (16 waves).
// xproj: [2][4096][1600] f32 pre-activations (x@Wih.T + b), row = b*T + t.
// Whh:   [2][1600][416] bf16.  hout: [4096][800] bf16, dir writes +dir*400.
__global__ void __launch_bounds__(512)
lstm_layer(const float* __restrict__ xproj,
           const __bf16* __restrict__ Whh,
           __bf16* __restrict__ hout) {
  __shared__ __bf16 hbuf[2][BB][KP0];   // ping-pong h state (52 KB)

  const int dir = blockIdx.x;
  const float*  xp = xproj + (size_t)dir * ROWS * G4;
  const __bf16* W  = Whh   + (size_t)dir * G4 * KP0;
  const int lane = threadIdx.x & 31;
  const int wave = threadIdx.x >> 5;

  for (int i = threadIdx.x; i < 2 * BB * KP0; i += blockDim.x)
    (&hbuf[0][0][0])[i] = (__bf16)0.0f;
  __syncthreads();

  float creg[4][8];                     // cell state lives in registers
#pragma unroll
  for (int uu = 0; uu < 4; ++uu)
#pragma unroll
    for (int r = 0; r < 8; ++r) creg[uu][r] = 0.0f;

  for (int s = 0; s < TT; ++s) {
    const int t = dir ? (TT - 1 - s) : s;
    const int pb = s & 1;               // read buffer; write pb^1
#pragma unroll
    for (int uu = 0; uu < 4; ++uu) {
      const int u = wave + uu * 16;     // 50 units: 2 m-tiles x 25 hidden-tiles
      if (u < 50) {
        const int m0 = (u / 25) << 4;
        const int n0 = (u % 25) << 4;
        v8f acc0, acc1, acc2, acc3;     // i, f, g, o gate tiles
#pragma unroll
        for (int r = 0; r < 8; ++r) {
          const int m = m0 + r + ((lane >> 4) << 3);
          const size_t base = ((size_t)m * TT + t) * G4 + n0 + (lane & 15);
          acc0[r] = xp[base];
          acc1[r] = xp[base + HH];
          acc2[r] = xp[base + 2 * HH];
          acc3[r] = xp[base + 3 * HH];
        }
        for (int k0 = 0; k0 < KP0; k0 += 32) {
          v16bf a = frag_A(&hbuf[pb][0][0], KP0, m0, k0);
          v16bf bi = frag_B(W, KP0, 0 * HH + n0, k0);
          acc0 = __builtin_amdgcn_wmma_f32_16x16x32_bf16(false, a, false, bi, (short)0, acc0, false, false);
          v16bf bf = frag_B(W, KP0, 1 * HH + n0, k0);
          acc1 = __builtin_amdgcn_wmma_f32_16x16x32_bf16(false, a, false, bf, (short)0, acc1, false, false);
          v16bf bg = frag_B(W, KP0, 2 * HH + n0, k0);
          acc2 = __builtin_amdgcn_wmma_f32_16x16x32_bf16(false, a, false, bg, (short)0, acc2, false, false);
          v16bf bo = frag_B(W, KP0, 3 * HH + n0, k0);
          acc3 = __builtin_amdgcn_wmma_f32_16x16x32_bf16(false, a, false, bo, (short)0, acc3, false, false);
        }
#pragma unroll
        for (int r = 0; r < 8; ++r) {
          const float ig = sigf(acc0[r]);
          const float fg = sigf(acc1[r]);
          const float gg = tanhf(acc2[r]);
          const float og = sigf(acc3[r]);
          const float cn = fg * creg[uu][r] + ig * gg;
          const float hn = og * tanhf(cn);
          creg[uu][r] = cn;
          const int m  = m0 + r + ((lane >> 4) << 3);
          const int hc = n0 + (lane & 15);
          const __bf16 hb = (__bf16)hn;
          hbuf[pb ^ 1][m][hc] = hb;                                  // exclusive owner
          hout[((size_t)m * TT + t) * LH + dir * HH + hc] = hb;
        }
      }
    }
    __syncthreads();                    // one barrier per timestep (ping-pong)
  }
}

// ---------------- s_h/s_m = tanh(feat) @ Wo halves ----------------------
__global__ void __launch_bounds__(256)
tanh_dot(const float* __restrict__ headf, const float* __restrict__ modf,
         const float* __restrict__ Wo, float* __restrict__ sh, float* __restrict__ sm) {
  __shared__ float r1[256], r2[256];
  const int row = blockIdx.x;
  const int tid = threadIdx.x;
  float a = 0.0f, b = 0.0f;
  for (int j = tid; j < MHH; j += 256) {
    a += tanhf(headf[(size_t)row * MHH + j]) * Wo[j];
    b += tanhf(modf[(size_t)row * MHH + j]) * Wo[MHH + j];
  }
  r1[tid] = a; r2[tid] = b;
  __syncthreads();
  for (int s = 128; s > 0; s >>= 1) {
    if (tid < s) { r1[tid] += r1[tid + s]; r2[tid] += r2[tid + s]; }
    __syncthreads();
  }
  if (tid == 0) { sh[row] = r1[0]; sm[row] = r2[0]; }
}

// ---------------- score[b,h,m] = s_h[b,h] + s_m[b,m] + bo ---------------
__global__ void score_outer(const float* __restrict__ sh, const float* __restrict__ sm,
                            const float* __restrict__ bo, float* __restrict__ out) {
  const int row = blockIdx.x;           // b*T + h
  const int b = row >> 7;               // /T
  const float base = sh[row] + bo[0];
  out[(size_t)row * TT + threadIdx.x] = base + sm[b * TT + threadIdx.x];
}

// ---------------------------------------------------------------------------
extern "C" void kernel_launch(void* const* d_in, const int* in_sizes, int n_in,
                              void* d_out, int out_size, void* d_ws, size_t ws_size,
                              hipStream_t stream) {
  (void)in_sizes; (void)n_in; (void)out_size;
  const int*   word  = (const int*)  d_in[0];
  const int*   pos   = (const int*)  d_in[2];
  const float* wemb  = (const float*)d_in[3];
  const float* pemb  = (const float*)d_in[4];
  const float* Wih0  = (const float*)d_in[5];
  const float* Whh0  = (const float*)d_in[6];
  const float* b0    = (const float*)d_in[7];
  const float* Wih1  = (const float*)d_in[8];
  const float* Whh1  = (const float*)d_in[9];
  const float* b1    = (const float*)d_in[10];
  const float* Wh    = (const float*)d_in[11];
  const float* bh    = (const float*)d_in[12];
  const float* Wm    = (const float*)d_in[13];
  const float* bm    = (const float*)d_in[14];
  const float* Wo    = (const float*)d_in[15];
  const float* bo    = (const float*)d_in[16];
  float* out = (float*)d_out;

  char* ws = (char*)d_ws;
  size_t off = 0;
  auto take = [&](size_t bytes) -> char* {
    char* p = ws + off;
    off += (bytes + 255) & ~(size_t)255;
    return p;
  };
  __bf16* x0    = (__bf16*)take((size_t)ROWS * KP0 * 2);
  __bf16* wih0b = (__bf16*)take((size_t)2 * G4 * KP0 * 2);
  __bf16* whh0b = (__bf16*)take((size_t)2 * G4 * KP0 * 2);
  __bf16* wih1b = (__bf16*)take((size_t)2 * G4 * LH * 2);
  __bf16* whh1b = (__bf16*)take((size_t)2 * G4 * KP0 * 2);
  __bf16* whb   = (__bf16*)take((size_t)MHH * LH * 2);
  __bf16* wmb   = (__bf16*)take((size_t)MHH * LH * 2);
  float*  xp0   = (float*) take((size_t)2 * ROWS * G4 * 4);
  float*  xp1   = (float*) take((size_t)2 * ROWS * G4 * 4);
  __bf16* h0b   = (__bf16*)take((size_t)ROWS * LH * 2);
  __bf16* h1b   = (__bf16*)take((size_t)ROWS * LH * 2);
  float*  headf = (float*) take((size_t)ROWS * MHH * 4);
  float*  modf  = (float*) take((size_t)ROWS * MHH * 4);
  float*  shv   = (float*) take((size_t)ROWS * 4);
  float*  smv   = (float*) take((size_t)ROWS * 4);
  if (off > ws_size) return;            // workspace too small: bail safely

  // 1. weights -> bf16 (pad K 400 -> 416 where needed)
  cvt_pad<<<2 * G4, 256, 0, stream>>>(Wih0, wih0b, 400, KP0);
  cvt_pad<<<2 * G4, 256, 0, stream>>>(Whh0, whh0b, 400, KP0);
  cvt_pad<<<2 * G4, 256, 0, stream>>>(Wih1, wih1b, LH, LH);
  cvt_pad<<<2 * G4, 256, 0, stream>>>(Whh1, whh1b, 400, KP0);
  cvt_pad<<<MHH,    256, 0, stream>>>(Wh, whb, LH, LH);
  cvt_pad<<<MHH,    256, 0, stream>>>(Wm, wmb, LH, LH);

  // 2. embedding gather + concat
  embed_concat<<<ROWS, 128, 0, stream>>>(word, pos, wemb, pemb, x0);

  // 3. layer-0 input projections (both directions): [4096,416]x[416,1600]
  {
    const int strips = (ROWS / 16) * (G4 / 64);
    const int grid   = (strips + 7) / 8;
    gemm_bf16_wmma<<<grid, 256, 0, stream>>>(x0, KP0, wih0b, KP0, b0,
                                             xp0, G4, ROWS, G4, KP0);
    gemm_bf16_wmma<<<grid, 256, 0, stream>>>(x0, KP0, wih0b + (size_t)G4 * KP0, KP0,
                                             b0 + G4, xp0 + (size_t)ROWS * G4, G4,
                                             ROWS, G4, KP0);
  }

  // 4. layer-0 recurrence (persistent, 2 workgroups = 2 directions)
  lstm_layer<<<2, 512, 0, stream>>>(xp0, whh0b, h0b);

  // 5. layer-1 input projections: [4096,800]x[800,1600]
  {
    const int strips = (ROWS / 16) * (G4 / 64);
    const int grid   = (strips + 7) / 8;
    gemm_bf16_wmma<<<grid, 256, 0, stream>>>(h0b, LH, wih1b, LH, b1,
                                             xp1, G4, ROWS, G4, LH);
    gemm_bf16_wmma<<<grid, 256, 0, stream>>>(h0b, LH, wih1b + (size_t)G4 * LH, LH,
                                             b1 + G4, xp1 + (size_t)ROWS * G4, G4,
                                             ROWS, G4, LH);
  }

  // 6. layer-1 recurrence
  lstm_layer<<<2, 512, 0, stream>>>(xp1, whh1b, h1b);

  // 7. head / modifier projections: [4096,800]x[800,512]
  {
    const int strips = (ROWS / 16) * (MHH / 64);
    const int grid   = (strips + 7) / 8;
    gemm_bf16_wmma<<<grid, 256, 0, stream>>>(h1b, LH, whb, LH, bh,
                                             headf, MHH, ROWS, MHH, LH);
    gemm_bf16_wmma<<<grid, 256, 0, stream>>>(h1b, LH, wmb, LH, bm,
                                             modf, MHH, ROWS, MHH, LH);
  }

  // 8. tanh-dot reductions, 9. outer-sum score
  tanh_dot<<<ROWS, 256, 0, stream>>>(headf, modf, Wo, shv, smv);
  score_outer<<<ROWS, TT, 0, stream>>>(shv, smv, bo, out);
}
